// TSModel_56513179681517
// MI455X (gfx1250) — matrix operand order repaired and checked by reading, not compiled, not run on hardware
//
#include <hip/hip_runtime.h>

typedef __attribute__((ext_vector_type(16))) _Float16 v16h;
typedef __attribute__((ext_vector_type(8)))  float    v8f;

#define B_TOT 512
#define T_LEN 1024
#define HID   64
#define GATES 256          // 4*HID
#define ROWS  16           // batch rows per workgroup
#define HSZ   (ROWS * HID) // one h1 buffer, in halves

// Fast bounded-domain activations: v_exp_f32 + v_rcp_f32
__device__ __forceinline__ float sig_f(float x) {
  return __builtin_amdgcn_rcpf(1.0f + __expf(-x));
}
__device__ __forceinline__ float tanh_f(float x) {
  return 1.0f - 2.0f * __builtin_amdgcn_rcpf(__expf(2.0f * x) + 1.0f);
}

// A-fragment (16x32 f16 chunk of h1 at column k0), ISA 7.12.2 layout.
__device__ __forceinline__ v16h load_a_frag(const _Float16* h1s, int ln, int hi, int k0) {
  union { v16h v; uint32_t u[8]; } r;
  const uint32_t* row = (const uint32_t*)(h1s + ln * HID + k0);
#pragma unroll
  for (int v = 0; v < 8; ++v) {
    int kp = (v < 4) ? (8 * hi + 2 * v) : (16 + 8 * hi + 2 * (v - 4));
    r.u[v] = row[kp >> 1];
  }
  return r.v;
}

__global__ __launch_bounds__(128) void lstm2_kernel(
    const float* __restrict__ x, const float* __restrict__ W1,
    const float* __restrict__ b1, const float* __restrict__ W2,
    const float* __restrict__ b2, float* __restrict__ out) {
  __shared__ float    xt_s[T_LEN * ROWS];   // 64 KB, transposed [t][m]
  __shared__ _Float16 h1_s[2 * HSZ];        // double-buffered h1, 4 KB
  __shared__ float    g2_s[ROWS * 4];
  __shared__ float    h2_s[ROWS];

  const int tid  = threadIdx.x;
  const int wave = tid >> 5;      // 0..3: owns ALL 4 gates for h-chunk 16*wave
  const int lane = tid & 31;
  const int ln   = lane & 15;
  const int hi   = lane >> 4;
  const int b0   = blockIdx.x * ROWS;

  // ---- preload whole x tile (contiguous 64 KB) into LDS, transposed ----
  {
    const float4* xg = (const float4*)(x + (size_t)b0 * T_LEN);
#pragma unroll 1
    for (int i4 = tid; i4 < (ROWS * T_LEN) / 4; i4 += 128) {
      float4 v = xg[i4];
      int g = i4 * 4;
#pragma unroll
      for (int j = 0; j < 4; ++j) {
        int gg = g + j;                       // gg = m*T + t
        xt_s[(gg & (T_LEN - 1)) * ROWS + (gg >> 10)] = ((const float*)&v)[j];
      }
    }
  }

  // ---- B fragments: 4 gate tiles per wave, n = g*64 + 16*wave + ln ----
  v16h bfrag[4][2];
  float wx[4], bb[4];
#pragma unroll
  for (int g = 0; g < 4; ++g) {
    int n = g * HID + 16 * wave + ln;
    wx[g] = W1[n];                 // W1 row 0 = x weights
    bb[g] = b1[n];
#pragma unroll
    for (int ch = 0; ch < 2; ++ch) {
#pragma unroll
      for (int e = 0; e < 16; ++e) {
        int k = 32 * ch + ((e < 8) ? (8 * hi + e) : (16 + 8 * hi + (e - 8)));       // K in 0..63
        bfrag[g][ch][e] = (_Float16)W1[(1 + k) * GATES + n];
      }
    }
  }

  // wave 0: layer-2 B fragments (W2 64x4 zero-padded to 64x16)
  v16h b2frag[2];
  float w2h = 0.f, b2v = 0.f;
#pragma unroll
  for (int ch = 0; ch < 2; ++ch)
#pragma unroll
    for (int e = 0; e < 16; ++e) {
      int k = 32 * ch + ((e < 8) ? (8 * hi + e) : (16 + 8 * hi + (e - 8)));
      b2frag[ch][e] = (ln < 4) ? (_Float16)W2[k * 4 + ln] : (_Float16)0.0f;
    }
  if (ln < 4) { w2h = W2[HID * 4 + ln]; b2v = b2[ln]; }

  // ---- state: c1 fully register-resident (8 cells/lane), c2/h2 in wave 0 ----
  float c1[8];
#pragma unroll
  for (int r = 0; r < 8; ++r) c1[r] = 0.f;
  float c2 = 0.f;
  if (tid < ROWS) h2_s[tid] = 0.f;
  ((uint4*)h1_s)[tid]       = make_uint4(0, 0, 0, 0);   // zero both buffers
  ((uint4*)h1_s)[tid + 128] = make_uint4(0, 0, 0, 0);
  __syncthreads();

  const int h_col = 16 * wave + ln;   // this lane's h column

#pragma unroll 1
  for (int t = 0; t < T_LEN; ++t) {
    const _Float16* h_rd = h1_s + (((t & 1) ^ 1) * HSZ);  // h1 after step t-1
    _Float16*       h_wr = h1_s + ((t & 1) * HSZ);

    v16h a0 = load_a_frag(h_rd, ln, hi, 0);
    v16h a1 = load_a_frag(h_rd, ln, hi, 32);

    // ---- wave 0: layer-2 for step t-1 reuses the SAME A fragments ----
    if (wave == 0 && t > 0) {
      v8f acc2;
#pragma unroll
      for (int r = 0; r < 8; ++r)
        acc2[r] = (ln < 4) ? (h2_s[r + 8 * hi] * w2h + b2v) : 0.f;
      acc2 = __builtin_amdgcn_wmma_f32_16x16x32_f16(false, a0, false, b2frag[0],
                                                    (short)0, acc2, false, false);
      acc2 = __builtin_amdgcn_wmma_f32_16x16x32_f16(false, a1, false, b2frag[1],
                                                    (short)0, acc2, false, false);
      if (ln < 4) {
#pragma unroll
        for (int r = 0; r < 8; ++r)
          g2_s[(r + 8 * hi) * 4 + ln] = acc2[r];
      }
      if (tid < ROWS) {   // intra-wave LDS: in-order, no barrier needed
        float gi = g2_s[tid * 4 + 0], gj = g2_s[tid * 4 + 1];
        float gf = g2_s[tid * 4 + 2], go = g2_s[tid * 4 + 3];
        c2 = c2 * sig_f(gf + 1.0f) + sig_f(gi) * tanh_f(gj);
        float h2 = tanh_f(c2) * sig_f(go);
        h2_s[tid] = h2;
        out[(b0 + tid) * T_LEN + (t - 1)] = h2;
      }
    }

    // ---- layer-1: all 4 gate tiles of this wave's h-chunk ----
    float xv[8];
    {
      const float* xrow = xt_s + t * ROWS + 8 * hi;
#pragma unroll
      for (int r = 0; r < 8; ++r) xv[r] = xrow[r];
    }
    v8f acc[4];
#pragma unroll
    for (int g = 0; g < 4; ++g) {
#pragma unroll
      for (int r = 0; r < 8; ++r)
        acc[g][r] = xv[r] * wx[g] + bb[g];
      acc[g] = __builtin_amdgcn_wmma_f32_16x16x32_f16(false, a0, false, bfrag[g][0],
                                                      (short)0, acc[g], false, false);
      acc[g] = __builtin_amdgcn_wmma_f32_16x16x32_f16(false, a1, false, bfrag[g][1],
                                                      (short)0, acc[g], false, false);
    }

    // ---- LSTM elementwise entirely on WMMA accumulators in registers ----
    // cell (m = r+8*hi, h = h_col): i=acc[0], j=acc[1], f=acc[2], o=acc[3]
#pragma unroll
    for (int r = 0; r < 8; ++r) {
      float c = c1[r] * sig_f(acc[2][r] + 1.0f) + sig_f(acc[0][r]) * tanh_f(acc[1][r]);
      c1[r] = c;
      h_wr[(r + 8 * hi) * HID + h_col] = (_Float16)(tanh_f(c) * sig_f(acc[3][r]));
    }
    __syncthreads();   // the ONLY barrier per step
  }

  // ---- final layer-2 for t = T_LEN-1 ----
  if (wave == 0) {
    const _Float16* h_rd = h1_s + (((T_LEN - 1) & 1) * HSZ);
    v16h a0 = load_a_frag(h_rd, ln, hi, 0);
    v16h a1 = load_a_frag(h_rd, ln, hi, 32);
    v8f acc2;
#pragma unroll
    for (int r = 0; r < 8; ++r)
      acc2[r] = (ln < 4) ? (h2_s[r + 8 * hi] * w2h + b2v) : 0.f;
    acc2 = __builtin_amdgcn_wmma_f32_16x16x32_f16(false, a0, false, b2frag[0],
                                                  (short)0, acc2, false, false);
    acc2 = __builtin_amdgcn_wmma_f32_16x16x32_f16(false, a1, false, b2frag[1],
                                                  (short)0, acc2, false, false);
    if (ln < 4) {
#pragma unroll
      for (int r = 0; r < 8; ++r)
        g2_s[(r + 8 * hi) * 4 + ln] = acc2[r];
    }
    if (tid < ROWS) {
      float gi = g2_s[tid * 4 + 0], gj = g2_s[tid * 4 + 1];
      float gf = g2_s[tid * 4 + 2], go = g2_s[tid * 4 + 3];
      c2 = c2 * sig_f(gf + 1.0f) + sig_f(gi) * tanh_f(gj);
      out[(b0 + tid) * T_LEN + (T_LEN - 1)] = tanh_f(c2) * sig_f(go);
    }
  }
}

extern "C" void kernel_launch(void* const* d_in, const int* in_sizes, int n_in,
                              void* d_out, int out_size, void* d_ws, size_t ws_size,
                              hipStream_t stream) {
  const float* x  = (const float*)d_in[0];   // [512,1024,1]
  const float* W1 = (const float*)d_in[1];   // [65,256]
  const float* b1 = (const float*)d_in[2];   // [256]
  const float* W2 = (const float*)d_in[3];   // [65,4]
  const float* b2 = (const float*)d_in[4];   // [4]
  float* out = (float*)d_out;                // [512,1024,1]
  (void)in_sizes; (void)n_in; (void)out_size; (void)d_ws; (void)ws_size;
  dim3 grid(B_TOT / ROWS), block(128);
  hipLaunchKernelGGL(lstm2_kernel, grid, block, 0, stream, x, W1, b1, W2, b2, out);
}
